// CustomDeepseekV2MoE_36524401885994
// MI455X (gfx1250) — matrix operand order repaired and checked by reading, not compiled, not run on hardware
//
#include <hip/hip_runtime.h>

// ---------------------------------------------------------------------------
// DeepSeek-V2 MoE for gfx1250 (MI455X), wave32 + v_wmma_f32_16x16x32_bf16.
// Sparse grouped-GEMM formulation of the dense reference:
//   - gate+grouped-topk builds per-expert token lists (top-6 of 16)
//   - routed experts: expert-major 16-token tiles, fused up->SwiGLU->down,
//     weights converted fp32->bf16 in registers, fp32 WMMA accumulate
//   - shared expert handled by the same kernel (SI=2816 = 2 halves of 1408)
//   - routed/shared contributions atomically accumulated into zeroed d_out
// ---------------------------------------------------------------------------

#define T_TOK 1024
#define DIM   2048          // hidden size D
#define NEXP  16            // experts
#define IDIM  1408          // moe intermediate I
#define DPAD  (DIM + 8)     // LDS row pad (keeps 16B align, staggers banks)
#define IPAD  (IDIM + 8)

typedef __bf16 bf16;
typedef __attribute__((ext_vector_type(16))) __bf16 v16bf;
typedef __attribute__((ext_vector_type(8)))  __bf16 v8bf;
typedef __attribute__((ext_vector_type(8)))  float  v8f;

// 16 contiguous fp32 -> bf16 WMMA B-fragment (64B = 4x float4 per lane)
__device__ inline v16bf load16_bf(const float* __restrict__ p) {
    const float4* q = (const float4*)p;
    float4 f0 = q[0], f1 = q[1], f2 = q[2], f3 = q[3];
    v16bf r;
    r[0]  = (bf16)f0.x; r[1]  = (bf16)f0.y; r[2]  = (bf16)f0.z; r[3]  = (bf16)f0.w;
    r[4]  = (bf16)f1.x; r[5]  = (bf16)f1.y; r[6]  = (bf16)f1.z; r[7]  = (bf16)f1.w;
    r[8]  = (bf16)f2.x; r[9]  = (bf16)f2.y; r[10] = (bf16)f2.z; r[11] = (bf16)f2.w;
    r[12] = (bf16)f3.x; r[13] = (bf16)f3.y; r[14] = (bf16)f3.z; r[15] = (bf16)f3.w;
    return r;
}

// A-fragment: two 16B LDS reads (this lane-half's K[16h..16h+7] and K[16h+8..16h+15])
__device__ inline v16bf load_a(const bf16* p0, const bf16* p1) {
    v8bf lo = *(const v8bf*)p0;
    v8bf hi = *(const v8bf*)p1;
    v16bf a;
#pragma unroll
    for (int j = 0; j < 8; ++j) { a[j] = lo[j]; a[j + 8] = hi[j]; }
    return a;
}

__device__ inline v8f zero8() {
    v8f c;
#pragma unroll
    for (int j = 0; j < 8; ++j) c[j] = 0.0f;
    return c;
}

// fast silu: one v_exp_f32 + one v_rcp_f32 (no IEEE division chain);
// rcp's ~1ulp error is invisible at bf16 storage precision
__device__ inline float fast_silu(float g) {
    return g * __builtin_amdgcn_rcpf(1.0f + __expf(-g));
}

// ---------------------------------------------------------------------------
// Kernel 1: zero the (atomically accumulated) output + expert counters
// ---------------------------------------------------------------------------
__global__ void init_kernel(float* __restrict__ out, int* __restrict__ counts) {
    size_t idx = (size_t)blockIdx.x * blockDim.x + threadIdx.x;
    if (idx < (size_t)T_TOK * DIM) out[idx] = 0.0f;
    if (idx < NEXP) counts[idx] = 0;
}

// ---------------------------------------------------------------------------
// Kernel 2: gating + grouped top-k routing. One wave32 per token.
// ---------------------------------------------------------------------------
__global__ void gate_topk_kernel(const float* __restrict__ x,
                                 const float* __restrict__ gw,
                                 int* __restrict__ counts,
                                 int* __restrict__ tok_list,
                                 float* __restrict__ wt_list) {
    const int lane = threadIdx.x & 31;
    const int wid  = threadIdx.x >> 5;
    const int t    = blockIdx.x * 4 + wid;

    float lg[NEXP];
    const float* xr = x + (size_t)t * DIM;
#pragma unroll 1
    for (int e = 0; e < NEXP; ++e) {
        const float* gr = gw + (size_t)e * DIM;
        float acc = 0.0f;
        for (int d = lane; d < DIM; d += 32) acc += xr[d] * gr[d];
#pragma unroll
        for (int off = 16; off; off >>= 1) acc += __shfl_xor(acc, off);
        lg[e] = acc;                       // xor-reduce leaves result in all lanes
    }
    if (lane != 0) return;

    // softmax over 16 experts
    float mx = lg[0];
    for (int e = 1; e < NEXP; ++e) mx = fmaxf(mx, lg[e]);
    float s[NEXP], sum = 0.0f;
    for (int e = 0; e < NEXP; ++e) { s[e] = __expf(lg[e] - mx); sum += s[e]; }
    float inv = 1.0f / sum;
    for (int e = 0; e < NEXP; ++e) s[e] *= inv;

    // group scores (4 groups of 4 consecutive experts), pick best 2 groups
    float gs[4];
    for (int g = 0; g < 4; ++g) {
        float m = s[4 * g];
        for (int j = 1; j < 4; ++j) m = fmaxf(m, s[4 * g + j]);
        gs[g] = m;
    }
    int g0 = 0;
    for (int g = 1; g < 4; ++g) if (gs[g] > gs[g0]) g0 = g;
    int g1 = -1;
    for (int g = 0; g < 4; ++g)
        if (g != g0 && (g1 < 0 || gs[g] > gs[g1])) g1 = g;

    // top-6 among the 8 allowed experts (first-index tie-break, like lax.top_k)
    float masked[NEXP];
    for (int e = 0; e < NEXP; ++e) {
        int g = e >> 2;
        masked[e] = (g == g0 || g == g1) ? s[e] : -1.0f;
    }
    int   ids[6];
    float wv[6];
    float wsum = 0.0f;
    for (int k = 0; k < 6; ++k) {
        int best = 0;
        for (int e = 1; e < NEXP; ++e) if (masked[e] > masked[best]) best = e;
        ids[k] = best; wv[k] = masked[best]; wsum += masked[best];
        masked[best] = -1.0f;
    }
    float invw = 1.0f / wsum;
    for (int k = 0; k < 6; ++k) {
        int e    = ids[k];
        int slot = atomicAdd(&counts[e], 1);
        tok_list[e * T_TOK + slot] = t;
        wt_list[e * T_TOK + slot]  = wv[k] * invw;
    }
}

// ---------------------------------------------------------------------------
// Kernel 3: fused SwiGLU FFN over a 16-token tile (256 threads = 8 waves).
//   ROUTED: blockIdx = e*64 + tile  (expert-major: same-expert tiles run
//           together so the 34.6MB/expert weight set is served from L2)
//   shared: blockIdx = tile*2 + si_half  (SI = 2 x 1408)
// WMMA fragment mapping (wave32, 16x16x32 bf16):
//   A: lane<16 holds row (lane&15) K[k0..k0+15];
//      lane>=16 K[k0+16..k0+31] (matches B's contiguous K order).
//   B: lane = column (lane&15); lane<16 K[0..15], lane>=16 K[16..31]
//      -> 16 contiguous fp32 per lane from the weight row, cvt to bf16.
//   C: lane<16 rows 0..7 (vgpr j), lane>=16 rows 8..15; col = lane&15.
// ---------------------------------------------------------------------------
template <bool ROUTED>
__launch_bounds__(256, 1)
__global__ void ffn_kernel(const float* __restrict__ x,
                           const float* __restrict__ Wg,
                           const float* __restrict__ Wu,
                           const float* __restrict__ Wd,
                           const int*   __restrict__ counts,
                           const int*   __restrict__ tok_list,
                           const float* __restrict__ wt_list,
                           float* __restrict__ out) {
    __shared__ bf16  xs[16][DPAD];    // gathered token tile (bf16)     ~64KB
    __shared__ bf16  act[16][IPAD];   // silu(g)*u intermediate tile    ~45KB
    __shared__ int   toks[16];
    __shared__ float wts[16];

    const int tid  = threadIdx.x;
    const int lane = tid & 31;
    const int wid  = tid >> 5;

    const float *gA, *uA, *dA;
    int ld_d, dcol;
    if (ROUTED) {
        const int e    = blockIdx.x >> 6;
        const int tile = blockIdx.x & 63;
        const int cnt  = counts[e];
        if (tile * 16 >= cnt) return;           // uniform early exit
        gA = Wg + (size_t)e * IDIM * DIM;
        uA = Wu + (size_t)e * IDIM * DIM;
        dA = Wd + (size_t)e * DIM * IDIM;
        ld_d = IDIM; dcol = 0;
        if (tid < 16) {
            int s  = tile * 16 + tid;
            int sc = s < cnt ? s : cnt - 1;     // pad with last valid token, w=0
            toks[tid] = tok_list[e * T_TOK + sc];
            wts[tid]  = (s < cnt) ? wt_list[e * T_TOK + sc] * 2.5f : 0.0f;
        }
    } else {
        const int tile = blockIdx.x >> 1;
        const int half = blockIdx.x & 1;
        gA = Wg + (size_t)half * IDIM * DIM;    // sw_gate rows [half*1408 ..)
        uA = Wu + (size_t)half * IDIM * DIM;
        dA = Wd; ld_d = 2 * IDIM; dcol = half * IDIM;
        if (tid < 16) { toks[tid] = tile * 16 + tid; wts[tid] = 1.0f; }
    }
    __syncthreads();

    // gather + fp32->bf16 convert the 16 token rows
    for (int idx = tid; idx < 16 * DIM; idx += 256) {
        int m = idx >> 11, d = idx & (DIM - 1);
        xs[m][d] = (bf16)x[(size_t)toks[m] * DIM + d];
    }
    __syncthreads();

    const int mrow = lane & 15;
    const int aoff = (lane & 16) ? 16 : 0;
    const int boff = (lane & 16) ? 16 : 0;
    const int mb   = (lane & 16) ? 8 : 0;
    const int ncol = lane & 15;

    // ---- stage 1: act = silu(x@Wg^T) * (x@Wu^T); I split 8 waves x 176 cols
    for (int nt = 0; nt < 11; ++nt) {
        const int n0 = wid * 176 + nt * 16;
        v8f cg = zero8(), cu = zero8();
        const float* gp = gA + (size_t)(n0 + ncol) * DIM + boff;
        const float* up = uA + (size_t)(n0 + ncol) * DIM + boff;
#pragma unroll 2
        for (int k0 = 0; k0 < DIM; k0 += 32) {
            v16bf a  = load_a(&xs[mrow][k0 + aoff], &xs[mrow][k0 + aoff + 8]);
            v16bf bg = load16_bf(gp + k0);
            v16bf bu = load16_bf(up + k0);
            cg = __builtin_amdgcn_wmma_f32_16x16x32_bf16(false, a, false, bg,
                                                         (short)0, cg, false, false);
            cu = __builtin_amdgcn_wmma_f32_16x16x32_bf16(false, a, false, bu,
                                                         (short)0, cu, false, false);
        }
#pragma unroll
        for (int j = 0; j < 8; ++j) {
            act[mb + j][n0 + ncol] = (bf16)(fast_silu(cg[j]) * cu[j]);
        }
    }
    __syncthreads();

    // hoist per-row token pointers / weights out of the stage-2 loop
    float wreg[8];
    size_t obase[8];
#pragma unroll
    for (int j = 0; j < 8; ++j) {
        wreg[j]  = wts[mb + j];
        obase[j] = (size_t)toks[mb + j] * DIM;
    }

    // ---- stage 2: out[tok] += w * (act @ Wd^T); D split 8 waves x 256 cols
    for (int nt = 0; nt < 16; ++nt) {
        const int n0 = wid * 256 + nt * 16;
        const int d  = n0 + ncol;
        v8f c = zero8();
        const float* dp = dA + (size_t)d * ld_d + dcol + boff;
#pragma unroll 2
        for (int k0 = 0; k0 < IDIM; k0 += 32) {
            v16bf a  = load_a(&act[mrow][k0 + aoff], &act[mrow][k0 + aoff + 8]);
            v16bf bd = load16_bf(dp + k0);
            c = __builtin_amdgcn_wmma_f32_16x16x32_bf16(false, a, false, bd,
                                                        (short)0, c, false, false);
        }
#pragma unroll
        for (int j = 0; j < 8; ++j) {
            unsafeAtomicAdd(&out[obase[j] + d], c[j] * wreg[j]);
        }
    }
}

// ---------------------------------------------------------------------------
extern "C" void kernel_launch(void* const* d_in, const int* in_sizes, int n_in,
                              void* d_out, int out_size, void* d_ws, size_t ws_size,
                              hipStream_t stream) {
    const float* x       = (const float*)d_in[0];  // [T, D]
    const float* gate_w  = (const float*)d_in[1];  // [E, D]
    const float* w_gate  = (const float*)d_in[2];  // [E, I, D]
    const float* w_up    = (const float*)d_in[3];  // [E, I, D]
    const float* w_down  = (const float*)d_in[4];  // [E, D, I]
    const float* sw_gate = (const float*)d_in[5];  // [2I, D]
    const float* sw_up   = (const float*)d_in[6];  // [2I, D]
    const float* sw_down = (const float*)d_in[7];  // [D, 2I]
    float* out = (float*)d_out;

    char* ws = (char*)d_ws;
    int*   counts   = (int*)ws;                            // 16 ints
    int*   tok_list = (int*)(ws + 256);                    // 16*1024 ints
    float* wt_list  = (float*)(ws + 256 + NEXP * T_TOK * 4);

    // zero output + counters
    init_kernel<<<(T_TOK * DIM + 255) / 256, 256, 0, stream>>>(out, counts);
    // routing
    gate_topk_kernel<<<T_TOK / 4, 128, 0, stream>>>(x, gate_w, counts,
                                                    tok_list, wt_list);
    // shared expert (64 tiles x 2 SI halves)
    ffn_kernel<false><<<64 * 2, 256, 0, stream>>>(x, sw_gate, sw_up, sw_down,
                                                  nullptr, nullptr, nullptr, out);
    // routed experts (16 experts x 64 tiles, expert-major)
    ffn_kernel<true><<<NEXP * 64, 256, 0, stream>>>(x, w_gate, w_up, w_down,
                                                    counts, tok_list, wt_list, out);
}
